// NeighborCooccurrenceEncoder_11836929868498
// MI455X (gfx1250) — matrix operand order repaired and checked by reading, not compiled, not run on hardware
//
#include <hip/hip_runtime.h>
#include <hip/hip_bf16.h>

// ---------------------------------------------------------------------------
// NeighborCooccurrenceEncoder for MI455X (gfx1250, wave32)
//
// Algorithm: counts c in [0,128] -> 129-entry lookup table
//   T[c][g] = sum_f relu(c*w1[f]+b1[f]) * w2[g][f] + b2[g]
// Kernel 1 builds T with v_wmma_f32_16x16x32_f16 (hi/lo f16 split for ~f32
// accuracy). Kernel 2 is the bandwidth-bound streamer: per-batch O(N^2)
// counting in LDS, then out[b,n,:] = T[c1] + T[c2] with non-temporal b128
// stores (write-once output; keep table/IDs in WGP$/L2).
// ---------------------------------------------------------------------------

typedef __attribute__((ext_vector_type(16))) _Float16 v16h;
typedef __attribute__((ext_vector_type(8)))  float    v8f;
typedef __attribute__((ext_vector_type(4)))  float    v4f;

#define PADDED_NODE_ID (-1)
#define NN 128   // neighbors per batch
#define FF 128   // feature dim
#define NC 129   // distinct count values 0..128

// ---------------------------------------------------------------------------
// Kernel 1: build T[129][128] with WMMA. 1 wave per 16x16 output tile.
// Tiles: 9 along c (rows 0..143, rows >128 discarded), 8 along g.
// ---------------------------------------------------------------------------
__global__ __launch_bounds__(32)
void NCE_build_table_wmma(const float* __restrict__ w1,
                          const float* __restrict__ b1,
                          const float* __restrict__ w2,
                          const float* __restrict__ b2,
                          float* __restrict__ T) {
    const int lane   = threadIdx.x;        // 0..31
    const int tile_m = blockIdx.x % 9;     // count-value tile
    const int tile_n = blockIdx.x / 9;     // feature tile
    const int g16    = lane >> 4;          // lane half (0/1)
    const int lm     = lane & 15;

    const float cval = (float)(tile_m * 16 + lm);  // A row = count value
    const int   gcol = tile_n * 16 + lm;           // B col = output feature

    v8f acc = {};
#pragma unroll
    for (int kstep = 0; kstep < 4; ++kstep) {
        v16h ah, al, bh, bl;
#pragma unroll
        for (int e = 0; e < 16; ++e) {
            // A 16x32 f16 layout (ISA 7.12.2): lanes 0-15 hold K=0..7,16..23;
            // lanes 16-31 hold K=8..15,24..31 ->  K = e + 8*g16 + (e>=8 ? 8 : 0)
            const int fa = kstep * 32 + e + 8 * g16 + ((e >= 8) ? 8 : 0);
            float x = cval * w1[fa] + b1[fa];
            x = x > 0.0f ? x : 0.0f;                 // relu
            const _Float16 xh = (_Float16)x;
            ah[e] = xh;
            al[e] = (_Float16)(x - (float)xh);       // residual for accuracy

            // B 32x16 f16 layout: lane = column N, same K order as A:
            // K = e + 8*g16 + (e>=8 ? 8 : 0)
            const int fb = kstep * 32 + e + 8 * g16 + ((e >= 8) ? 8 : 0);
            const float y = w2[gcol * FF + fb];      // w2[g][f] row-major
            const _Float16 yh = (_Float16)y;
            bh[e] = yh;
            bl[e] = (_Float16)(y - (float)yh);
        }
        // D = Ah*Bh + Ah*Bl + Al*Bh + C  (~f32-accurate product)
        acc = __builtin_amdgcn_wmma_f32_16x16x32_f16(false, ah, false, bh,
                                                     (short)0, acc, false, false);
        acc = __builtin_amdgcn_wmma_f32_16x16x32_f16(false, ah, false, bl,
                                                     (short)0, acc, false, false);
        acc = __builtin_amdgcn_wmma_f32_16x16x32_f16(false, al, false, bh,
                                                     (short)0, acc, false, false);
    }

    const float bias = b2[gcol];
    // C/D layout: VGPR j holds M = j + 8*(lane/16), N = lane%16
#pragma unroll
    for (int j = 0; j < 8; ++j) {
        const int crow = tile_m * 16 + j + 8 * g16;
        if (crow < NC) T[crow * FF + gcol] = acc[j] + bias;
    }
}

// ---------------------------------------------------------------------------
// Kernel 2: bandwidth-bound streamer. 256 threads (8 waves), 8 batches/WG.
// ---------------------------------------------------------------------------
__global__ __launch_bounds__(256)
void NCE_encode_stream(const int* __restrict__ src,
                       const int* __restrict__ dst,
                       const float* __restrict__ T,
                       float* __restrict__ out_src,
                       float* __restrict__ out_dst) {
    __shared__ int s_src[NN];
    __shared__ int s_dst[NN];
    __shared__ int s_cnt[4][NN];   // css, csd, cdd, cds

    const int tid = threadIdx.x;
    const int b0  = blockIdx.x * 8;

    for (int bi = 0; bi < 8; ++bi) {
        const int b = b0 + bi;

        if (tid < NN) {
            // IDs are read exactly once -> non-temporal load
            s_src[tid] = __builtin_nontemporal_load(src + b * NN + tid);
            s_dst[tid] = __builtin_nontemporal_load(dst + b * NN + tid);
        }
        __syncthreads();

        if (tid < NN) {
            const int s = s_src[tid];
            const int d = s_dst[tid];
            int css = 0, csd = 0, cdd = 0, cds = 0;
#pragma unroll 4
            for (int j = 0; j < NN; ++j) {
                const int sj = s_src[j];
                const int dj = s_dst[j];
                css += (sj == s);   // #src == src[n]
                csd += (dj == s);   // #dst == src[n]
                cdd += (dj == d);   // #dst == dst[n]
                cds += (sj == d);   // #src == dst[n]
            }
            if (s == PADDED_NODE_ID) { css = 0; csd = 0; }  // freq -> 0 == T[0] path
            if (d == PADDED_NODE_ID) { cdd = 0; cds = 0; }
            s_cnt[0][tid] = css; s_cnt[1][tid] = csd;
            s_cnt[2][tid] = cdd; s_cnt[3][tid] = cds;
        }
        __syncthreads();

        // out[b,n,f] = T[c1][f] + T[c2][f]; 32 lanes cover 128 floats as v4f,
        // 8 row-groups in flight, 16 iterations. Output is write-once ->
        // non-temporal (streaming) stores; table loads stay regular-temporal
        // so T remains resident in WGP$/L2.
        const int f = (tid & 31) * 4;
        for (int n = tid >> 5; n < NN; n += 8) {
            const int c0 = s_cnt[0][n], c1 = s_cnt[1][n];
            const int c2 = s_cnt[2][n], c3 = s_cnt[3][n];
            const v4f t0 = *(const v4f*)(T + c0 * FF + f);
            const v4f t1 = *(const v4f*)(T + c1 * FF + f);
            const v4f t2 = *(const v4f*)(T + c2 * FF + f);
            const v4f t3 = *(const v4f*)(T + c3 * FF + f);
            const v4f os = t0 + t1;
            const v4f od = t2 + t3;
            const size_t idx = ((size_t)b * NN + n) * FF + f;
            __builtin_nontemporal_store(os, (v4f*)(out_src + idx));
            __builtin_nontemporal_store(od, (v4f*)(out_dst + idx));
        }
        __syncthreads();   // protect s_src/s_cnt before next batch
    }
}

// ---------------------------------------------------------------------------
// Launch
// ---------------------------------------------------------------------------
extern "C" void kernel_launch(void* const* d_in, const int* in_sizes, int n_in,
                              void* d_out, int out_size, void* d_ws, size_t ws_size,
                              hipStream_t stream) {
    const int*   src = (const int*)d_in[0];     // [B,128] int32
    const int*   dst = (const int*)d_in[1];     // [B,128] int32
    const float* w1  = (const float*)d_in[2];   // [128]
    const float* b1  = (const float*)d_in[3];   // [128]
    const float* w2  = (const float*)d_in[4];   // [128,128] (out,in)
    const float* b2  = (const float*)d_in[5];   // [128]

    const int B = in_sizes[0] / NN;             // 2048

    float* T = (float*)d_ws;                    // 129*128 f32 = 66 KB scratch
    float* out_src = (float*)d_out;             // [B,128,128]
    float* out_dst = out_src + (size_t)B * NN * FF;

    // 9x8 = 72 tiles, one wave32 each; 12 WMMAs per tile.
    NCE_build_table_wmma<<<72, 32, 0, stream>>>(w1, b1, w2, b2, T);
    // 256 workgroups x 8 waves; same stream so T is ready before use.
    NCE_encode_stream<<<B / 8, 256, 0, stream>>>(src, dst, T, out_src, out_dst);
}
